// FlashGQAAttention_60705067762221
// MI455X (gfx1250) — compile-verified, matrix operand order checked by reading, not run on hardware
//
#include <hip/hip_runtime.h>
#include <hip/hip_bf16.h>
#include <math.h>

// ---------------------------------------------------------------------------
// Problem constants (match reference)
// ---------------------------------------------------------------------------
#define B_SZ   2
#define T_SEQ  4096
#define D_MOD  1024
#define HQ_N   16
#define HKV_N  4
#define HD_N   64
#define WIN_R  256
#define M_TOT  (B_SZ * T_SEQ)          // 8192 token rows
#define N_QKV  (HQ_N*HD_N + 2*HKV_N*HD_N) // 1536

typedef __attribute__((ext_vector_type(16))) _Float16 v16h;
typedef __attribute__((ext_vector_type(8)))  _Float16 v8h;
typedef __attribute__((ext_vector_type(8)))  float    v8f;

// ---------------------------------------------------------------------------
// WMMA helpers
// ---------------------------------------------------------------------------
__device__ __forceinline__ v8f wmma_f16(v16h a, v16h b, v8f c) {
  return __builtin_amdgcn_wmma_f32_16x16x32_f16(
      /*neg_a=*/false, a, /*neg_b=*/false, b,
      /*c_mod=*/(short)0, c, /*reuse_a=*/false, /*reuse_b=*/false);
}

// Load one 16x32 f16 fragment row/col slice.
// Fragment half-layout (wave32): for lane L, lo = (L>=16)*8:
//   halves 0..7  -> K = kbase + lo + h        (contiguous 16B)
//   halves 8..15 -> K = kbase + 16 + lo + h   (contiguous 16B)
// `p` points at element K = kbase of this lane's row/col; `lo` = 0 or 8.
__device__ __forceinline__ v16h load_frag(const _Float16* __restrict__ p, int lo) {
  v8h a = *(const v8h*)(p + lo);
  v8h b = *(const v8h*)(p + 16 + lo);
  v16h r;
#pragma unroll
  for (int i = 0; i < 8; ++i) { r[i] = a[i]; r[i + 8] = b[i]; }
  return r;
}

// ---------------------------------------------------------------------------
// Pre-pass: f32 -> f16 convert, and f32 [K][N] -> f16 [N][K] weight transpose
// ---------------------------------------------------------------------------
__global__ void __launch_bounds__(256)
cvt_f32_to_f16_kernel(const float* __restrict__ src, _Float16* __restrict__ dst, int n) {
  int i = blockIdx.x * blockDim.x + threadIdx.x;
  if (i < n) dst[i] = (_Float16)src[i];
}

__global__ void __launch_bounds__(256)
transpose_w_kernel(const float* __restrict__ src, _Float16* __restrict__ dst, int K, int N) {
  int i = blockIdx.x * blockDim.x + threadIdx.x;
  if (i < K * N) {
    int k = i / N, n = i - k * N;
    dst[(size_t)n * K + k] = (_Float16)src[i];
  }
}

// ---------------------------------------------------------------------------
// Fused QKV GEMM: C[8192 x 1536] = xh[8192 x 1024] * W, per-wave 64x64 tile.
// Epilogue scatters into qh (row-major), kh (row-major), vT (d-major,
// token-contiguous) so the attention kernel gets contiguous b128 fragments.
// ---------------------------------------------------------------------------
__global__ void __launch_bounds__(256)
gemm_qkv_kernel(const _Float16* __restrict__ xh,
                const _Float16* __restrict__ wqT,
                const _Float16* __restrict__ wkT,
                const _Float16* __restrict__ wvT,
                const float* __restrict__ bq,
                const float* __restrict__ bk,
                const float* __restrict__ bv,
                _Float16* __restrict__ qh,
                _Float16* __restrict__ kh,
                _Float16* __restrict__ vT) {
  const int lane = threadIdx.x & 31;
  const int l16  = lane & 15;
  const int hi8  = (lane >> 4) * 8;
  const int gid  = blockIdx.x * (blockDim.x >> 5) + (threadIdx.x >> 5);
  const int NT   = N_QKV / 64;            // 24 col tiles
  const int mt   = gid / NT;
  const int nt   = gid - mt * NT;
  const int m0   = mt * 64;

  // Select weight region (tile never crosses a region boundary: 1024,1280 % 64==0)
  const _Float16* Bt; int reg, crel0;
  if (nt < 16)      { Bt = wqT + (size_t)(nt * 64) * D_MOD;        reg = 0; crel0 = nt * 64; }
  else if (nt < 20) { Bt = wkT + (size_t)((nt - 16) * 64) * D_MOD; reg = 1; crel0 = (nt - 16) * 64; }
  else              { Bt = wvT + (size_t)((nt - 20) * 64) * D_MOD; reg = 2; crel0 = (nt - 20) * 64; }

  v8f acc[4][4];
#pragma unroll
  for (int mi = 0; mi < 4; ++mi)
#pragma unroll
    for (int ni = 0; ni < 4; ++ni)
#pragma unroll
      for (int r = 0; r < 8; ++r) acc[mi][ni][r] = 0.0f;

  for (int kb = 0; kb < D_MOD; kb += 32) {
    v16h af[4], bf[4];
#pragma unroll
    for (int mi = 0; mi < 4; ++mi)
      af[mi] = load_frag(xh + (size_t)(m0 + mi * 16 + l16) * D_MOD + kb, hi8);
#pragma unroll
    for (int ni = 0; ni < 4; ++ni)
      bf[ni] = load_frag(Bt + (size_t)(ni * 16 + l16) * D_MOD + kb, hi8);
#pragma unroll
    for (int mi = 0; mi < 4; ++mi)
#pragma unroll
      for (int ni = 0; ni < 4; ++ni)
        acc[mi][ni] = wmma_f16(af[mi], bf[ni], acc[mi][ni]);
  }

  if (reg == 0) {
#pragma unroll
    for (int ni = 0; ni < 4; ++ni) {
      int c = crel0 + ni * 16 + l16;
      float bias = bq[c];
#pragma unroll
      for (int mi = 0; mi < 4; ++mi)
#pragma unroll
        for (int r = 0; r < 8; ++r) {
          int m = m0 + mi * 16 + r + hi8;
          qh[(size_t)m * (HQ_N * HD_N) + c] = (_Float16)(acc[mi][ni][r] + bias);
        }
    }
  } else if (reg == 1) {
#pragma unroll
    for (int ni = 0; ni < 4; ++ni) {
      int c = crel0 + ni * 16 + l16;
      float bias = bk[c];
#pragma unroll
      for (int mi = 0; mi < 4; ++mi)
#pragma unroll
        for (int r = 0; r < 8; ++r) {
          int m = m0 + mi * 16 + r + hi8;
          kh[(size_t)m * (HKV_N * HD_N) + c] = (_Float16)(acc[mi][ni][r] + bias);
        }
    }
  } else {
#pragma unroll
    for (int ni = 0; ni < 4; ++ni) {
      int c = crel0 + ni * 16 + l16;     // c = hkv*64 + d
      float bias = bv[c];
#pragma unroll
      for (int mi = 0; mi < 4; ++mi)
#pragma unroll
        for (int r = 0; r < 8; ++r) {
          int m = m0 + mi * 16 + r + hi8;
          int bb = m >> 12;              // / 4096
          int t  = m & (T_SEQ - 1);
          vT[((size_t)bb * (HKV_N * HD_N) + c) * T_SEQ + t] = (_Float16)(acc[mi][ni][r] + bias);
        }
    }
  }
}

// ---------------------------------------------------------------------------
// Sliding-window GQA flash attention. One wave per (batch, q-head, 16-query
// block). Computes S^T = K_tile x Q^T so softmax stats are per-lane (one
// shuffle). Key tiles are processed in PAIRS (32 keys/iter): the two S^T
// tiles fill the P^T B-fragment exactly (halves 0..7 = tile0, 8..15 = tile1)
// and V^T fragments are full 16x32 contiguous loads -> no zero padding, one
// online-softmax update per 32 keys. A single zero-padded tile is peeled
// first when the window tile count is odd.
// ---------------------------------------------------------------------------
__global__ void __launch_bounds__(256)
attn_kernel(const _Float16* __restrict__ qh,
            const _Float16* __restrict__ kh,
            const _Float16* __restrict__ vT,
            _Float16* __restrict__ yh) {
  const int lane = threadIdx.x & 31;
  const int l16  = lane & 15;
  const int lo   = (lane >> 4) * 8;
  const int wid  = blockIdx.x * (blockDim.x >> 5) + (threadIdx.x >> 5);

  const int qb  = wid & (T_SEQ / 16 - 1);   // 256 query blocks
  const int h   = (wid >> 8) & (HQ_N - 1);
  const int b   = wid >> 12;
  const int i0  = qb * 16;
  const int hkv = h >> 2;                   // G = HQ/HKV = 4

  // Q^T B-fragments (lane = query column, halves = d), two K-steps of 32
  const _Float16* qbase = qh + ((size_t)(b * T_SEQ + i0 + l16)) * (HQ_N * HD_N) + h * HD_N;
  const v16h qB0 = load_frag(qbase, lo);
  const v16h qB1 = load_frag(qbase + 32, lo);

  const _Float16* krow = kh + ((size_t)(b * T_SEQ + l16)) * (HKV_N * HD_N) + hkv * HD_N;
  const size_t    vrow0 = ((size_t)(b * (HKV_N * HD_N) + hkv * HD_N + l16)) * T_SEQ;

  v8f o[4];
#pragma unroll
  for (int dt = 0; dt < 4; ++dt)
#pragma unroll
    for (int r = 0; r < 8; ++r) o[dt][r] = 0.0f;

  float m_run = -1e30f, l_run = 0.0f;
  const int qpos = i0 + l16;

  int lo_t = i0 - WIN_R;       if (lo_t < 0) lo_t = 0;
  int hi_t = i0 + 15 + WIN_R;  if (hi_t > T_SEQ - 1) hi_t = T_SEQ - 1;

  int ntile = ((hi_t - lo_t) >> 4) + 1;
  int t0 = lo_t;

  // ---- peel one single (zero-padded) tile if the tile count is odd ----
  if (ntile & 1) {
    const _Float16* kbase = krow + (size_t)t0 * (HKV_N * HD_N);
    v16h kA0 = load_frag(kbase, lo);
    v16h kA1 = load_frag(kbase + 32, lo);

    v8f s;
#pragma unroll
    for (int r = 0; r < 8; ++r) s[r] = 0.0f;
    s = wmma_f16(kA0, qB0, s);
    s = wmma_f16(kA1, qB1, s);

    float tmax = -INFINITY;
#pragma unroll
    for (int r = 0; r < 8; ++r) {
      float sv = s[r] * 0.125f;               // 1/sqrt(64)
      int kpos = t0 + r + lo;                 // C row = r + 8*(lane>=16)
      int dlt  = kpos - qpos;
      if (dlt < -WIN_R || dlt > WIN_R) sv = -INFINITY;
      s[r] = sv;
      tmax = fmaxf(tmax, sv);
    }
    tmax = fmaxf(tmax, __shfl_xor(tmax, 16, 32));
    float m_new = fmaxf(m_run, tmax);
    float sc    = __expf(m_run - m_new);

    float psum = 0.0f;
    v16h pB;
#pragma unroll
    for (int r = 0; r < 8; ++r) {
      float p = __expf(s[r] - m_new);
      psum += p;
      pB[r]     = (_Float16)p;
      pB[r + 8] = (_Float16)0.0f;             // zero-pad K = 16..31
    }
    psum += __shfl_xor(psum, 16, 32);
    l_run = l_run * sc + psum;
    m_run = m_new;

#pragma unroll
    for (int dt = 0; dt < 4; ++dt)
#pragma unroll
      for (int r = 0; r < 8; ++r) o[dt][r] *= sc;

#pragma unroll
    for (int dt = 0; dt < 4; ++dt) {
      const _Float16* vbase = vT + vrow0 + (size_t)dt * 16 * T_SEQ + t0 + lo;
      v8h vl = *(const v8h*)vbase;
      v16h vA;
#pragma unroll
      for (int i = 0; i < 8; ++i) { vA[i] = vl[i]; vA[i + 8] = (_Float16)0.0f; }
      o[dt] = wmma_f16(vA, pB, o[dt]);
    }
    t0 += 16;
  }

  // ---- main loop: pairs of key tiles (32 keys per iteration) ----
  for (; t0 < hi_t; t0 += 32) {
    const _Float16* kbase0 = krow + (size_t)t0 * (HKV_N * HD_N);
    const _Float16* kbase1 = kbase0 + (size_t)16 * (HKV_N * HD_N);
    v16h kA0a = load_frag(kbase0, lo);
    v16h kA0b = load_frag(kbase0 + 32, lo);
    v16h kA1a = load_frag(kbase1, lo);
    v16h kA1b = load_frag(kbase1 + 32, lo);

    v8f s0, s1;
#pragma unroll
    for (int r = 0; r < 8; ++r) { s0[r] = 0.0f; s1[r] = 0.0f; }
    s0 = wmma_f16(kA0a, qB0, s0);
    s0 = wmma_f16(kA0b, qB1, s0);
    s1 = wmma_f16(kA1a, qB0, s1);
    s1 = wmma_f16(kA1b, qB1, s1);

    float tmax = -INFINITY;
#pragma unroll
    for (int r = 0; r < 8; ++r) {
      float sv0 = s0[r] * 0.125f;
      float sv1 = s1[r] * 0.125f;
      int kp0 = t0 + r + lo;
      int kp1 = kp0 + 16;
      int d0 = kp0 - qpos;
      int d1 = kp1 - qpos;
      if (d0 < -WIN_R || d0 > WIN_R) sv0 = -INFINITY;
      if (d1 < -WIN_R || d1 > WIN_R) sv1 = -INFINITY;
      s0[r] = sv0; s1[r] = sv1;
      tmax = fmaxf(tmax, fmaxf(sv0, sv1));
    }
    tmax = fmaxf(tmax, __shfl_xor(tmax, 16, 32));
    float m_new = fmaxf(m_run, tmax);
    float sc    = __expf(m_run - m_new);

    float psum = 0.0f;
    v16h pB;                                  // full 32-key P^T fragment
#pragma unroll
    for (int r = 0; r < 8; ++r) {
      float p0 = __expf(s0[r] - m_new);       // K = lo + r     (keys t0   + r+8hi)
      float p1 = __expf(s1[r] - m_new);       // K = 16+lo + r  (keys t0+16+ r+8hi)
      psum += p0 + p1;
      pB[r]     = (_Float16)p0;
      pB[r + 8] = (_Float16)p1;
    }
    psum += __shfl_xor(psum, 16, 32);
    l_run = l_run * sc + psum;
    m_run = m_new;

#pragma unroll
    for (int dt = 0; dt < 4; ++dt)
#pragma unroll
      for (int r = 0; r < 8; ++r) o[dt][r] *= sc;

    // O^T += V^T(16d x 32keys) x P^T(32keys x 16q): full-K fragments
#pragma unroll
    for (int dt = 0; dt < 4; ++dt) {
      v16h vA = load_frag(vT + vrow0 + (size_t)dt * 16 * T_SEQ + t0, lo);
      o[dt] = wmma_f16(vA, pB, o[dt]);
    }
  }

  const float inv = 1.0f / l_run;
#pragma unroll
  for (int dt = 0; dt < 4; ++dt) {
    v8h ov;
#pragma unroll
    for (int r = 0; r < 8; ++r) ov[r] = (_Float16)(o[dt][r] * inv);
    *(v8h*)(yh + ((size_t)(b * T_SEQ + i0 + l16)) * (HQ_N * HD_N) + h * HD_N + dt * 16 + lo) = ov;
  }
}

// ---------------------------------------------------------------------------
// Output projection: out[8192 x 1024] = yh * wo + bo (f32 result)
// ---------------------------------------------------------------------------
__global__ void __launch_bounds__(256)
gemm_out_kernel(const _Float16* __restrict__ yh,
                const _Float16* __restrict__ woT,
                const float* __restrict__ bo,
                float* __restrict__ out) {
  const int lane = threadIdx.x & 31;
  const int l16  = lane & 15;
  const int hi8  = (lane >> 4) * 8;
  const int gid  = blockIdx.x * (blockDim.x >> 5) + (threadIdx.x >> 5);
  const int NT   = D_MOD / 64;            // 16 col tiles
  const int mt   = gid / NT;
  const int nt   = gid - mt * NT;
  const int m0   = mt * 64;
  const int n0   = nt * 64;

  v8f acc[4][4];
#pragma unroll
  for (int mi = 0; mi < 4; ++mi)
#pragma unroll
    for (int ni = 0; ni < 4; ++ni)
#pragma unroll
      for (int r = 0; r < 8; ++r) acc[mi][ni][r] = 0.0f;

  for (int kb = 0; kb < HQ_N * HD_N; kb += 32) {
    v16h af[4], bf[4];
#pragma unroll
    for (int mi = 0; mi < 4; ++mi)
      af[mi] = load_frag(yh + (size_t)(m0 + mi * 16 + l16) * (HQ_N * HD_N) + kb, hi8);
#pragma unroll
    for (int ni = 0; ni < 4; ++ni)
      bf[ni] = load_frag(woT + (size_t)(n0 + ni * 16 + l16) * (HQ_N * HD_N) + kb, hi8);
#pragma unroll
    for (int mi = 0; mi < 4; ++mi)
#pragma unroll
      for (int ni = 0; ni < 4; ++ni)
        acc[mi][ni] = wmma_f16(af[mi], bf[ni], acc[mi][ni]);
  }

#pragma unroll
  for (int ni = 0; ni < 4; ++ni) {
    int c = n0 + ni * 16 + l16;
    float bias = bo[c];
#pragma unroll
    for (int mi = 0; mi < 4; ++mi)
#pragma unroll
      for (int r = 0; r < 8; ++r) {
        int m = m0 + mi * 16 + r + hi8;
        out[(size_t)m * D_MOD + c] = acc[mi][ni][r] + bias;
      }
  }
}

// ---------------------------------------------------------------------------
// Host-side launcher (graph-capture safe: launches only, all on `stream`)
// ---------------------------------------------------------------------------
extern "C" void kernel_launch(void* const* d_in, const int* in_sizes, int n_in,
                              void* d_out, int out_size, void* d_ws, size_t ws_size,
                              hipStream_t stream) {
  const float* x  = (const float*)d_in[0];
  const float* wq = (const float*)d_in[1];
  const float* bq = (const float*)d_in[2];
  const float* wk = (const float*)d_in[3];
  const float* bk = (const float*)d_in[4];
  const float* wv = (const float*)d_in[5];
  const float* bv = (const float*)d_in[6];
  const float* wo = (const float*)d_in[7];
  const float* bo = (const float*)d_in[8];
  float* out = (float*)d_out;

  // Workspace layout (f16 elements)
  _Float16* ws = (_Float16*)d_ws;
  const size_t SZ_X  = (size_t)M_TOT * D_MOD;          // 8192*1024
  const size_t SZ_K  = (size_t)M_TOT * (HKV_N * HD_N); // 8192*256
  _Float16* xh  = ws;                    // x in f16
  _Float16* qh  = xh  + SZ_X;            // Q rows [token][hq*64+d]
  _Float16* yh  = qh  + SZ_X;            // attention output rows
  _Float16* wqT = yh  + SZ_X;            // [1024][1024]
  _Float16* woT = wqT + (size_t)D_MOD * D_MOD;
  _Float16* wkT = woT + (size_t)D_MOD * D_MOD;     // [256][1024]
  _Float16* wvT = wkT + (size_t)(HKV_N * HD_N) * D_MOD;
  _Float16* kh  = wvT + (size_t)(HKV_N * HD_N) * D_MOD; // K rows
  _Float16* vTt = kh  + SZ_K;            // V transposed [b][hkv*64+d][t]

  // 1) converts / transposes
  {
    int n = (int)SZ_X;
    cvt_f32_to_f16_kernel<<<(n + 255) / 256, 256, 0, stream>>>(x, xh, n);
    transpose_w_kernel<<<(D_MOD * D_MOD + 255) / 256, 256, 0, stream>>>(wq, wqT, D_MOD, D_MOD);
    transpose_w_kernel<<<(D_MOD * (HKV_N * HD_N) + 255) / 256, 256, 0, stream>>>(wk, wkT, D_MOD, HKV_N * HD_N);
    transpose_w_kernel<<<(D_MOD * (HKV_N * HD_N) + 255) / 256, 256, 0, stream>>>(wv, wvT, D_MOD, HKV_N * HD_N);
    transpose_w_kernel<<<(D_MOD * D_MOD + 255) / 256, 256, 0, stream>>>(wo, woT, D_MOD, D_MOD);
  }

  // 2) fused QKV projection: (8192/64) * (1536/64) = 3072 waves, 8 waves/block
  gemm_qkv_kernel<<<(M_TOT / 64) * (N_QKV / 64) / 8, 256, 0, stream>>>(
      xh, wqT, wkT, wvT, bq, bk, bv, qh, kh, vTt);

  // 3) sliding-window GQA attention: B*HQ*(T/16) = 8192 waves, 8 waves/block
  attn_kernel<<<(B_SZ * HQ_N * (T_SEQ / 16)) / 8, 256, 0, stream>>>(qh, kh, vTt, yh);

  // 4) output projection: (8192/64)*(1024/64) = 2048 waves, 8 waves/block
  gemm_out_kernel<<<(M_TOT / 64) * (D_MOD / 64) / 8, 256, 0, stream>>>(yh, woT, bo, out);
}
